// NSAAttention_39084202394160
// MI455X (gfx1250) — matrix operand order repaired and checked by reading, not compiled, not run on hardware
//
#include <hip/hip_runtime.h>
#include <hip/hip_bf16.h>

#define B_  2
#define L_  2048
#define C_  256
#define H_  8
#define HD_ 32
#define BS_ 16
#define NB_ (L_/BS_)   /* 128 */
#define SK_ 16
#define SCALE_ 0.17677669529663687f  /* 1/sqrt(32) */
#define NEG_  -10000.0f

typedef _Float16 v16h __attribute__((ext_vector_type(16)));
typedef float    v8f  __attribute__((ext_vector_type(8)));

// Global (address_space(1)) element types so loads/stores lower to
// global_load_*/global_store_* (LOADcnt/STOREcnt only) instead of FLAT
// (which also ties up DScnt and the LDS port on CDNA5).
typedef __attribute__((address_space(1))) _Float16 g16;
typedef __attribute__((address_space(1))) float    g32;

// ---------------------------------------------------------------------------
// WMMA fragment helpers (layouts per CDNA5 ISA 7.12.2, wave32)
// A (16x32 f16): lane L holds row M=L&15.
//   lanes 0-15 : halves 0..7 -> K=0..7,  halves 8..15 -> K=16..23
//   lanes 16-31: halves 0..7 -> K=8..15, halves 8..15 -> K=24..31
// ---------------------------------------------------------------------------
static __device__ inline v16h load_a_frag_g(const g16* __restrict__ src, int ld, int lane) {
  int m  = lane & 15;
  int hi = (lane >> 4) & 1;
  const g16* row = src + (size_t)m * ld;
  int kA = hi ? 8  : 0;
  int kB = hi ? 24 : 16;
  v16h a;
#pragma unroll
  for (int e = 0; e < 8; ++e) a[e]     = row[kA + e];
#pragma unroll
  for (int e = 0; e < 8; ++e) a[8 + e] = row[kB + e];
  return a;
}

// A fragment sourced from LDS-resident fp32 (softmax weights)
static __device__ inline v16h load_a_frag_f32(const float* src, int ld, int lane) {
  int m  = lane & 15;
  int hi = (lane >> 4) & 1;
  const float* row = src + (size_t)m * ld;
  int kA = hi ? 8  : 0;
  int kB = hi ? 24 : 16;
  v16h a;
#pragma unroll
  for (int e = 0; e < 8; ++e) a[e]     = (_Float16)row[kA + e];
#pragma unroll
  for (int e = 0; e < 8; ++e) a[8 + e] = (_Float16)row[kB + e];
  return a;
}

// B (32x16 f16) loaded from N-major ("transposed") storage Bt[n][k].
// lane L: N = L&15, K = koff + e (e<8), koff + 16 + (e-8) (e>=8), koff = (L<16 ? 0 : 8).
static __device__ inline v16h load_b_frag_g(const g16* __restrict__ bt, int ld, int lane) {
  int n    = lane & 15;
  int koff = (lane & 16) ? 8 : 0;
  const g16* p = bt + (size_t)n * ld + koff;
  v16h b;
#pragma unroll
  for (int e = 0; e < 8; ++e) { b[e] = p[e]; b[8 + e] = p[16 + e]; }
  return b;
}

static __device__ inline v8f wmma16(v16h a, v16h b, v8f c) {
  return __builtin_amdgcn_wmma_f32_16x16x32_f16(false, a, false, b, (short)0, c, false, false);
}

// ---------------------------------------------------------------------------
// Pack kernels
// ---------------------------------------------------------------------------
__global__ void f32_to_f16_kernel(const float* __restrict__ src_, _Float16* __restrict__ dst_, int n) {
  const g32* src = (const g32*)src_;
  g16* dst = (g16*)dst_;
  int i = blockIdx.x * blockDim.x + threadIdx.x;
  if (i < n) dst[i] = (_Float16)src[i];
}

__global__ void transpose_w_kernel(const float* __restrict__ src_, _Float16* __restrict__ dst_) {
  const g32* src = (const g32*)src_;
  g16* dst = (g16*)dst_;
  int i = blockIdx.x * blockDim.x + threadIdx.x;  // over C_*C_
  if (i >= C_ * C_) return;
  int k = i / C_, n = i % C_;
  dst[(size_t)n * C_ + k] = (_Float16)src[(size_t)k * C_ + n];
}

// ---------------------------------------------------------------------------
// QKV projection: Q=query@Wq+bq etc., stored head-major [B][H][L][HD] f16
// grid = B*L/16 row tiles, block = 256 (8 waves, 6 col-tiles each: 48 = 3*16)
// ---------------------------------------------------------------------------
__global__ void __launch_bounds__(256) proj_qkv_kernel(
    const _Float16* __restrict__ xq, const _Float16* __restrict__ xk, const _Float16* __restrict__ xv,
    const _Float16* __restrict__ wqT, const _Float16* __restrict__ wkT, const _Float16* __restrict__ wvT,
    const float* __restrict__ bq, const float* __restrict__ bk, const float* __restrict__ bv,
    _Float16* __restrict__ qh, _Float16* __restrict__ kh, _Float16* __restrict__ vh)
{
  int r0   = blockIdx.x * 16;
  int wave = threadIdx.x >> 5;
  int lane = threadIdx.x & 31;
  int m = lane & 15, hi = lane >> 4;
  const g16* xs[3] = {(const g16*)xq, (const g16*)xk, (const g16*)xv};
  const g16* ws[3] = {(const g16*)wqT, (const g16*)wkT, (const g16*)wvT};
  const g32* bs[3] = {(const g32*)bq, (const g32*)bk, (const g32*)bv};
  g16*       os[3] = {(g16*)qh, (g16*)kh, (g16*)vh};

  for (int t = 0; t < 6; ++t) {
    int tileid = wave * 6 + t;         // 0..47
    int mat = tileid >> 4;
    int n0  = (tileid & 15) << 4;
    const g16* X  = xs[mat];
    const g16* WT = ws[mat];
    v8f c = {};
    for (int k0 = 0; k0 < C_; k0 += 32) {
      v16h a = load_a_frag_g(X + (size_t)r0 * C_ + k0, C_, lane);
      v16h b = load_b_frag_g(WT + (size_t)n0 * C_ + k0, C_, lane);
      c = wmma16(a, b, c);
    }
    int ncol = n0 + m;
    float bias = bs[mat][ncol];
    int h = ncol >> 5, d = ncol & 31;
    g16* O = os[mat];
#pragma unroll
    for (int i = 0; i < 8; ++i) {
      int row = r0 + i + hi * 8;           // global row in [0, B*L)
      int bb = row / L_, l = row % L_;
      O[(((size_t)bb * H_ + h) * L_ + l) * HD_ + d] = (_Float16)(c[i] + bias);
    }
  }
}

// ---------------------------------------------------------------------------
// Gate: sigmoid(query @ Wg + bg), one thread per (b,l)
// ---------------------------------------------------------------------------
__global__ void gate_kernel(const float* __restrict__ query_, const float* __restrict__ Wg_,
                            const float* __restrict__ bg_, float* __restrict__ gate_) {
  const g32* query = (const g32*)query_;
  const g32* Wg = (const g32*)Wg_;
  const g32* bg = (const g32*)bg_;
  g32* gate = (g32*)gate_;
  int i = blockIdx.x * blockDim.x + threadIdx.x;
  if (i >= B_ * L_) return;
  const g32* x = query + (size_t)i * C_;
#pragma unroll
  for (int j = 0; j < 3; ++j) {
    float acc = bg[j];
    for (int k = 0; k < C_; ++k) acc += x[k] * Wg[k * 3 + j];
    gate[(size_t)i * 3 + j] = 1.0f / (1.0f + __expf(-acc));
  }
}

// ---------------------------------------------------------------------------
// Compressed K/V: Kc[bh][n][d] f16, VcT[bh][d][n] f16
// ---------------------------------------------------------------------------
__global__ void compress_kernel(const _Float16* __restrict__ kh_, const _Float16* __restrict__ vh_,
                                const float* __restrict__ WKc_, const float* __restrict__ WVc_,
                                const float* __restrict__ Wpe_,
                                _Float16* __restrict__ kc_, _Float16* __restrict__ vcT_) {
  const g16* kh = (const g16*)kh_;
  const g16* vh = (const g16*)vh_;
  const g32* WKc = (const g32*)WKc_;
  const g32* WVc = (const g32*)WVc_;
  const g32* Wpe = (const g32*)Wpe_;
  g16* kc = (g16*)kc_;
  g16* vcT = (g16*)vcT_;
  int t = blockIdx.x * blockDim.x + threadIdx.x;
  if (t >= B_ * H_ * NB_ * HD_) return;
  int d  = t & (HD_ - 1);
  int n  = (t >> 5) & (NB_ - 1);
  int bh = t >> 12;                 // / (NB_*HD_)
  float ak = 0.f, av = 0.f;
#pragma unroll
  for (int s = 0; s < BS_; ++s) {
    float pe = Wpe[s * C_ + d];
    size_t idx = ((size_t)bh * L_ + n * BS_ + s) * HD_ + d;
    ak += WKc[s] * ((float)kh[idx] + pe);
    av += WVc[s] * ((float)vh[idx] + pe);
  }
  kc[((size_t)bh * NB_ + n) * HD_ + d]  = (_Float16)ak;
  vcT[((size_t)bh * HD_ + d) * NB_ + n] = (_Float16)av;
}

// ---------------------------------------------------------------------------
// Compressed attention: per wave one (b,h,q-tile). 4 waves/block (LDS budget).
// ---------------------------------------------------------------------------
__global__ void __launch_bounds__(128) cmp_attn_kernel(const _Float16* __restrict__ qh_,
                                                       const _Float16* __restrict__ kc_,
                                                       const _Float16* __restrict__ vcT_,
                                                       float* __restrict__ o_cmp_) {
  const g16* qh = (const g16*)qh_;
  const g16* kc = (const g16*)kc_;
  const g16* vcT = (const g16*)vcT_;
  g32* o_cmp = (g32*)o_cmp_;
  __shared__ float sS[4][16][NB_];   // 32 KB
  int wave = threadIdx.x >> 5, lane = threadIdx.x & 31;
  int task = blockIdx.x * 4 + wave;       // 2048 tasks
  int qt = task & 127, bh = task >> 7;
  int q0 = qt << 4;
  int m = lane & 15, hi = lane >> 4;

  v16h a = load_a_frag_g(qh + ((size_t)bh * L_ + q0) * HD_, HD_, lane);
  for (int nt = 0; nt < NB_ / 16; ++nt) {
    v16h b = load_b_frag_g(kc + ((size_t)bh * NB_ + nt * 16) * HD_, HD_, lane);
    v8f c = {};
    c = wmma16(a, b, c);
#pragma unroll
    for (int i = 0; i < 8; ++i) sS[wave][i + hi * 8][nt * 16 + m] = c[i] * SCALE_;
  }
  __syncthreads();
  if (lane < 16) {
    int r = lane;
    float mx = -1e30f;
    for (int k = 0; k < NB_; ++k) mx = fmaxf(mx, sS[wave][r][k]);
    float sum = 0.f;
    for (int k = 0; k < NB_; ++k) { float e = __expf(sS[wave][r][k] - mx); sS[wave][r][k] = e; sum += e; }
    float inv = 1.0f / sum;
    for (int k = 0; k < NB_; ++k) sS[wave][r][k] *= inv;
  }
  __syncthreads();
  for (int dt = 0; dt < 2; ++dt) {
    v8f c = {};
    for (int k0 = 0; k0 < NB_; k0 += 32) {
      v16h aw = load_a_frag_f32(&sS[wave][0][0] + k0, NB_, lane);
      v16h b  = load_b_frag_g(vcT + ((size_t)bh * HD_ + dt * 16) * NB_ + k0, NB_, lane);
      c = wmma16(aw, b, c);
    }
#pragma unroll
    for (int i = 0; i < 8; ++i)
      o_cmp[((size_t)bh * L_ + q0 + i + hi * 8) * HD_ + dt * 16 + m] = c[i];
  }
}

// ---------------------------------------------------------------------------
// Selection attention: stream all key tiles through WMMA, keep top-16/row.
// 8 waves/block, one (b,h,q-tile) per wave.
// ---------------------------------------------------------------------------
__global__ void __launch_bounds__(256) sel_attn_kernel(const _Float16* __restrict__ qh_,
                                                       const _Float16* __restrict__ kh_,
                                                       const _Float16* __restrict__ vh_,
                                                       float* __restrict__ o_slc_) {
  const g16* qh = (const g16*)qh_;
  const g16* kh = (const g16*)kh_;
  const g16* vh = (const g16*)vh_;
  g32* o_slc = (g32*)o_slc_;
  __shared__ float sS[8][16][16];
  __shared__ float sTV[8][16][SK_];
  __shared__ int   sTI[8][16][SK_];
  int wave = threadIdx.x >> 5, lane = threadIdx.x & 31;
  int task = blockIdx.x * 8 + wave;
  int qt = task & 127, bh = task >> 7;
  int q0 = qt << 4;
  int m = lane & 15, hi = lane >> 4;

  v16h a = load_a_frag_g(qh + ((size_t)bh * L_ + q0) * HD_, HD_, lane);
  float minv = -1e30f; int minp = 0;
  if (lane < 16) {
#pragma unroll
    for (int j = 0; j < SK_; ++j) { sTV[wave][lane][j] = -1e30f; sTI[wave][lane][j] = 0; }
  }
  for (int kt = 0; kt < L_ / 16; ++kt) {
    v16h b = load_b_frag_g(kh + ((size_t)bh * L_ + kt * 16) * HD_, HD_, lane);
    v8f c = {};
    c = wmma16(a, b, c);
#pragma unroll
    for (int i = 0; i < 8; ++i) sS[wave][i + hi * 8][m] = c[i] * SCALE_;
    if (lane < 16) {
      int r = lane;
      for (int j = 0; j < 16; ++j) {
        float v = sS[wave][r][j];
        if (v > minv) {
          sTV[wave][r][minp] = v;
          sTI[wave][r][minp] = kt * 16 + j;
          minv = sTV[wave][r][0]; minp = 0;
          for (int u = 1; u < SK_; ++u) {
            float tv = sTV[wave][r][u];
            if (tv < minv) { minv = tv; minp = u; }
          }
        }
      }
    }
  }
  if (lane < 16) {
    int r = lane;
    float mx = -1e30f;
    for (int j = 0; j < SK_; ++j) mx = fmaxf(mx, sTV[wave][r][j]);
    float w[SK_]; float sum = 0.f;
#pragma unroll
    for (int j = 0; j < SK_; ++j) { w[j] = __expf(sTV[wave][r][j] - mx); sum += w[j]; }
    float inv = 1.0f / sum;
    float acc[HD_];
#pragma unroll
    for (int d = 0; d < HD_; ++d) acc[d] = 0.f;
    for (int j = 0; j < SK_; ++j) {
      const g16* vr = vh + ((size_t)bh * L_ + sTI[wave][r][j]) * HD_;
      float wj = w[j] * inv;
#pragma unroll
      for (int d = 0; d < HD_; ++d) acc[d] += wj * (float)vr[d];
    }
    g32* orow = o_slc + ((size_t)bh * L_ + q0 + r) * HD_;
#pragma unroll
    for (int d = 0; d < HD_; ++d) orow[d] = acc[d];
  }
}

// ---------------------------------------------------------------------------
// Window attention: banded |i-j|<=32; <=5 key tiles per query tile.
// ---------------------------------------------------------------------------
__global__ void __launch_bounds__(256) win_attn_kernel(const _Float16* __restrict__ qh_,
                                                       const _Float16* __restrict__ kh_,
                                                       const _Float16* __restrict__ vh_,
                                                       float* __restrict__ o_win_) {
  const g16* qh = (const g16*)qh_;
  const g16* kh = (const g16*)kh_;
  const g16* vh = (const g16*)vh_;
  g32* o_win = (g32*)o_win_;
  __shared__ float sS[8][16][80];   // 40 KB
  int wave = threadIdx.x >> 5, lane = threadIdx.x & 31;
  int task = blockIdx.x * 8 + wave;
  int qt = task & 127, bh = task >> 7;
  int q0 = qt << 4;
  int m = lane & 15, hi = lane >> 4;

  int kstart = q0 - 32; if (kstart < 0) kstart = 0;
  int kend   = q0 + 48; if (kend > L_) kend = L_;
  int nt = (kend - kstart) >> 4;

  v16h a = load_a_frag_g(qh + ((size_t)bh * L_ + q0) * HD_, HD_, lane);
  for (int t = 0; t < nt; ++t) {
    v16h b = load_b_frag_g(kh + ((size_t)bh * L_ + kstart + t * 16) * HD_, HD_, lane);
    v8f c = {};
    c = wmma16(a, b, c);
#pragma unroll
    for (int i = 0; i < 8; ++i) {
      int qi = q0 + i + hi * 8;
      int kj = kstart + t * 16 + m;
      int df = qi - kj; if (df < 0) df = -df;
      sS[wave][i + hi * 8][t * 16 + m] = (df <= 32) ? c[i] * SCALE_ : NEG_;
    }
  }
  if (lane < 16) {
    int r = lane;
    int nk = nt * 16;
    float mx = -1e30f;
    for (int k = 0; k < nk; ++k) mx = fmaxf(mx, sS[wave][r][k]);
    float sum = 0.f;
    for (int k = 0; k < nk; ++k) { float e = __expf(sS[wave][r][k] - mx); sS[wave][r][k] = e; sum += e; }
    float inv = 1.0f / sum;
    float acc[HD_];
#pragma unroll
    for (int d = 0; d < HD_; ++d) acc[d] = 0.f;
    for (int k = 0; k < nk; ++k) {
      float wgt = sS[wave][r][k] * inv;
      const g16* vr = vh + ((size_t)bh * L_ + kstart + k) * HD_;
#pragma unroll
      for (int d = 0; d < HD_; ++d) acc[d] += wgt * (float)vr[d];
    }
    g32* orow = o_win + ((size_t)bh * L_ + q0 + r) * HD_;
#pragma unroll
    for (int d = 0; d < HD_; ++d) orow[d] = acc[d];
  }
}

// ---------------------------------------------------------------------------
// Combine gated branches into [B][L][C] f16
// ---------------------------------------------------------------------------
__global__ void combine_kernel(const float* __restrict__ o_cmp_, const float* __restrict__ o_slc_,
                               const float* __restrict__ o_win_, const float* __restrict__ gate_,
                               _Float16* __restrict__ xcomb_) {
  const g32* o_cmp = (const g32*)o_cmp_;
  const g32* o_slc = (const g32*)o_slc_;
  const g32* o_win = (const g32*)o_win_;
  const g32* gate = (const g32*)gate_;
  g16* xcomb = (g16*)xcomb_;
  int t = blockIdx.x * blockDim.x + threadIdx.x;
  if (t >= B_ * L_ * C_) return;
  int d = t & 31;
  int h = (t >> 5) & 7;
  int c = t & (C_ - 1);
  int l = (t >> 8) & (L_ - 1);
  int b = t >> 19;
  size_t hidx = (((size_t)b * H_ + h) * L_ + l) * HD_ + d;
  const g32* g = gate + ((size_t)b * L_ + l) * 3;
  float v = g[0] * o_cmp[hidx] + g[1] * o_slc[hidx] + g[2] * o_win[hidx];
  xcomb[((size_t)b * L_ + l) * C_ + c] = (_Float16)v;
}

// ---------------------------------------------------------------------------
// Output projection: out = xcomb @ Wo + bo (fp32 out)
// ---------------------------------------------------------------------------
__global__ void __launch_bounds__(256) out_proj_kernel(const _Float16* __restrict__ x_,
                                                       const _Float16* __restrict__ woT_,
                                                       const float* __restrict__ bo_,
                                                       float* __restrict__ out_) {
  const g16* x = (const g16*)x_;
  const g16* woT = (const g16*)woT_;
  const g32* bo = (const g32*)bo_;
  g32* out = (g32*)out_;
  int r0   = blockIdx.x * 16;
  int wave = threadIdx.x >> 5;
  int lane = threadIdx.x & 31;
  int m = lane & 15, hi = lane >> 4;
  for (int t = 0; t < 2; ++t) {
    int n0 = (wave * 2 + t) * 16;
    v8f c = {};
    for (int k0 = 0; k0 < C_; k0 += 32) {
      v16h a = load_a_frag_g(x + (size_t)r0 * C_ + k0, C_, lane);
      v16h b = load_b_frag_g(woT + (size_t)n0 * C_ + k0, C_, lane);
      c = wmma16(a, b, c);
    }
    float bias = bo[n0 + m];
#pragma unroll
    for (int i = 0; i < 8; ++i)
      out[((size_t)(r0 + i + hi * 8)) * C_ + n0 + m] = c[i] + bias;
  }
}

// ---------------------------------------------------------------------------
extern "C" void kernel_launch(void* const* d_in, const int* in_sizes, int n_in,
                              void* d_out, int out_size, void* d_ws, size_t ws_size,
                              hipStream_t stream) {
  const float* query = (const float*)d_in[0];
  const float* key_  = (const float*)d_in[1];
  const float* value = (const float*)d_in[2];
  const float* Wq = (const float*)d_in[3];  const float* bq = (const float*)d_in[4];
  const float* Wk = (const float*)d_in[5];  const float* bk = (const float*)d_in[6];
  const float* Wv = (const float*)d_in[7];  const float* bv = (const float*)d_in[8];
  const float* Wo = (const float*)d_in[9];  const float* bo = (const float*)d_in[10];
  const float* WKc = (const float*)d_in[11];
  const float* WVc = (const float*)d_in[12];
  const float* Wpe = (const float*)d_in[13];
  const float* Wg  = (const float*)d_in[14]; const float* bg = (const float*)d_in[15];
  float* out = (float*)d_out;

  const size_t BLC = (size_t)B_ * L_ * C_;      // 1,048,576
  char* p = (char*)d_ws;
  auto alloc = [&](size_t bytes) -> char* {
    char* r = p;
    p += (bytes + 255) & ~((size_t)255);
    return r;
  };
  _Float16* xq_h = (_Float16*)alloc(BLC * 2);
  _Float16* xk_h = (_Float16*)alloc(BLC * 2);
  _Float16* xv_h = (_Float16*)alloc(BLC * 2);
  _Float16* wqT  = (_Float16*)alloc((size_t)C_ * C_ * 2);
  _Float16* wkT  = (_Float16*)alloc((size_t)C_ * C_ * 2);
  _Float16* wvT  = (_Float16*)alloc((size_t)C_ * C_ * 2);
  _Float16* woT  = (_Float16*)alloc((size_t)C_ * C_ * 2);
  _Float16* qh   = (_Float16*)alloc(BLC * 2);
  _Float16* kh   = (_Float16*)alloc(BLC * 2);
  _Float16* vh   = (_Float16*)alloc(BLC * 2);
  _Float16* kc   = (_Float16*)alloc((size_t)B_ * H_ * NB_ * HD_ * 2);
  _Float16* vcT  = (_Float16*)alloc((size_t)B_ * H_ * NB_ * HD_ * 2);
  _Float16* xcomb= (_Float16*)alloc(BLC * 2);
  float* gate  = (float*)alloc((size_t)B_ * L_ * 3 * 4);
  float* o_cmp = (float*)alloc(BLC * 4);
  float* o_slc = (float*)alloc(BLC * 4);
  float* o_win = (float*)alloc(BLC * 4);

  const int TPB = 256;
  int nBLC = (int)BLC;
  // pack activations + weights to f16 (weights transposed to N-major)
  f32_to_f16_kernel<<<(nBLC + TPB - 1) / TPB, TPB, 0, stream>>>(query, xq_h, nBLC);
  f32_to_f16_kernel<<<(nBLC + TPB - 1) / TPB, TPB, 0, stream>>>(key_,  xk_h, nBLC);
  f32_to_f16_kernel<<<(nBLC + TPB - 1) / TPB, TPB, 0, stream>>>(value, xv_h, nBLC);
  int nW = C_ * C_;
  transpose_w_kernel<<<(nW + TPB - 1) / TPB, TPB, 0, stream>>>(Wq, wqT);
  transpose_w_kernel<<<(nW + TPB - 1) / TPB, TPB, 0, stream>>>(Wk, wkT);
  transpose_w_kernel<<<(nW + TPB - 1) / TPB, TPB, 0, stream>>>(Wv, wvT);
  transpose_w_kernel<<<(nW + TPB - 1) / TPB, TPB, 0, stream>>>(Wo, woT);

  // QKV projections (WMMA)
  proj_qkv_kernel<<<B_ * L_ / 16, 256, 0, stream>>>(xq_h, xk_h, xv_h, wqT, wkT, wvT,
                                                    bq, bk, bv, qh, kh, vh);
  // gate
  gate_kernel<<<(B_ * L_ + TPB - 1) / TPB, TPB, 0, stream>>>(query, Wg, bg, gate);
  // compressed K/V
  int nC = B_ * H_ * NB_ * HD_;
  compress_kernel<<<(nC + TPB - 1) / TPB, TPB, 0, stream>>>(kh, vh, WKc, WVc, Wpe, kc, vcT);
  // three attention branches (WMMA)
  cmp_attn_kernel<<<(B_ * H_ * (L_ / 16)) / 4, 128, 0, stream>>>(qh, kc, vcT, o_cmp);
  sel_attn_kernel<<<(B_ * H_ * (L_ / 16)) / 8, 256, 0, stream>>>(qh, kh, vh, o_slc);
  win_attn_kernel<<<(B_ * H_ * (L_ / 16)) / 8, 256, 0, stream>>>(qh, kh, vh, o_win);
  // gated combine + output projection (WMMA)
  combine_kernel<<<(nBLC + TPB - 1) / TPB, TPB, 0, stream>>>(o_cmp, o_slc, o_win, gate, xcomb);
  out_proj_kernel<<<B_ * L_ / 16, 256, 0, stream>>>(xcomb, woT, bo, out);
}
